// FlatNetRec_18665927868996
// MI455X (gfx1250) — matrix operand order repaired and checked by reading, not compiled, not run on hardware
//
#include <hip/hip_runtime.h>

typedef _Float16 v8h  __attribute__((ext_vector_type(8)));
typedef _Float16 v16h __attribute__((ext_vector_type(16)));
typedef float    v8f  __attribute__((ext_vector_type(8)));

#define WAVES_PER_WG 4
#define LDS_HALVES_PER_WAVE 4096   // 8 KB per wave: bufA (4KB) + bufB (4KB)

// Pre-swizzled weight-fragment offsets in d_ws (units: halves).
// Each fragment = 512 halves (16 per lane x 32 lanes) in WMMA 16-bit B layout.
#define FRAG_CODE 0        // 8 ntiles * 3 ksteps = 24 frags  (96 x128)
#define FRAG_H1   12288    // 4 * 4 = 16 frags                (128x 64)
#define FRAG_H2   20480    // 2 * 2 = 4 frags                 (64 x 32)
#define FRAG_H3   22528    // 1 frag                          (32 x 16)
#define FRAG_H4   23040    // 1 frag, zero-padded             (16 x  3 -> 32x16)

// ---------------------------------------------------------------------------
// Prep kernel: convert fp32 weights -> f16 fragments for the CDNA5 WMMA
// 16-bit B operand in contiguous-k order (as the A loads): lane l holds col n = n0+(l&15); K values
// k0 + 16*(l>>4) + i for i=0..15, stored contiguously per lane.
// ---------------------------------------------------------------------------
__global__ void prep_weights(const float* __restrict__ Wc, const float* __restrict__ W1,
                             const float* __restrict__ W2, const float* __restrict__ W3,
                             const float* __restrict__ W4, _Float16* __restrict__ frag) {
    int f = blockIdx.x;          // fragment id 0..45
    int t = threadIdx.x;         // 0..511 -> (lane, i)
    int lane = t >> 4, i = t & 15;
    const float* W; int KT, Ksrc, Nsrc, base, fl;
    if (f < 24)       { W = Wc; KT = 3; Ksrc = 96;  Nsrc = 128; base = FRAG_CODE; fl = f; }
    else if (f < 40)  { W = W1; KT = 4; Ksrc = 128; Nsrc = 64;  base = FRAG_H1;   fl = f - 24; }
    else if (f < 44)  { W = W2; KT = 2; Ksrc = 64;  Nsrc = 32;  base = FRAG_H2;   fl = f - 40; }
    else if (f == 44) { W = W3; KT = 1; Ksrc = 32;  Nsrc = 16;  base = FRAG_H3;   fl = 0; }
    else              { W = W4; KT = 1; Ksrc = 16;  Nsrc = 3;   base = FRAG_H4;   fl = 0; }
    int nt = fl / KT, kt = fl % KT;
    int n  = nt * 16 + (lane & 15);
    int kk = kt * 32 + 16 * (lane >> 4) + i;
    float v = (kk < Ksrc && n < Nsrc) ? W[kk * Nsrc + n] : 0.0f;
    frag[base + fl * 512 + t] = (_Float16)v;
}

// ---------------------------------------------------------------------------
// One dense layer: out = act(in @ W + b) via v_wmma_f32_16x16x32_f16.
// A-fragments are hoisted and loaded once; n-tiles sweep over them.
// in  : LDS tile [16 x KDIM] f16, row stride KDIM
// out : LDS tile [16 x NDIM] f16, row stride OSTR
// ACT : 0 = relu, 1 = leaky(0.2)
// ---------------------------------------------------------------------------
template <int KDIM, int NDIM, int OSTR, int ACT>
__device__ __forceinline__ void mlp_layer(const _Float16* in, _Float16* out,
                                          const _Float16* __restrict__ frag,
                                          const float* __restrict__ bias, int lane) {
    const int hi = lane >> 4, ln = lane & 15;
    constexpr int KT = KDIM / 32, NT = NDIM / 16;
    // A layout: lane<16 holds K [k0..k0+15] of row ln;
    // lane>=16 holds K [k0+16..k0+31].
    v16h afr[KT];
#pragma unroll
    for (int kt = 0; kt < KT; ++kt) {
        const _Float16* ap = in + ln * KDIM + kt * 32 + hi * 16;
        v8h a0 = *(const v8h*)ap;
        v8h a1 = *(const v8h*)(ap + 8);
        afr[kt] = __builtin_shufflevector(a0, a1, 0, 1, 2, 3, 4, 5, 6, 7,
                                          8, 9, 10, 11, 12, 13, 14, 15);
    }
#pragma unroll
    for (int nt = 0; nt < NT; ++nt) {
        float bn = bias[nt * 16 + ln];
        v8f c = {bn, bn, bn, bn, bn, bn, bn, bn};
#pragma unroll
        for (int kt = 0; kt < KT; ++kt) {
            v16h b = *(const v16h*)(frag + (nt * KT + kt) * 512 + lane * 16);
            c = __builtin_amdgcn_wmma_f32_16x16x32_f16(false, afr[kt], false, b,
                                                       (short)0, c, false, false);
        }
        // C layout: VGPR g -> row g + 8*hi, col = ln
#pragma unroll
        for (int g = 0; g < 8; ++g) {
            float x = c[g];
            if (ACT == 0) x = fmaxf(x, 0.0f);
            else          x = fmaxf(x, 0.2f * x);
            out[(g + hi * 8) * OSTR + nt * 16 + ln] = (_Float16)x;
        }
    }
}

// ---------------------------------------------------------------------------
// Fused kernel: one wave owns 16 blocks (one 16-row WMMA tile) end-to-end.
// Waves are fully independent (private LDS slices, same-wave DS ordering),
// so no barriers are required.
// ---------------------------------------------------------------------------
__global__ __launch_bounds__(128) void flatnet_kernel(
    const float* __restrict__ pgi,
    const float* __restrict__ W_lift, const float* __restrict__ b_lift,
    const float* __restrict__ b_code, const float* __restrict__ b_h1,
    const float* __restrict__ b_h2,   const float* __restrict__ b_h3,
    const float* __restrict__ b_h4,
    const _Float16* __restrict__ frag,
    float* __restrict__ out) {
    __shared__ __align__(16) _Float16 lds[WAVES_PER_WG * LDS_HALVES_PER_WAVE];
    const int lane = threadIdx.x & 31;
    const int wid  = threadIdx.x >> 5;
    _Float16* bufA = lds + wid * LDS_HALVES_PER_WAVE;  // v-tile / h1-out / h3-out
    _Float16* bufB = bufA + 2048;                      // code-out / h2-out
    const int tile = blockIdx.x * WAVES_PER_WG + wid;
    const int blockBase = tile * 16;

    // ---- stage A: per-point lift (3->32), ring-group max, then leaky ----
    // leaky_relu(.,0.2) is monotonic increasing, so
    //   max_p leaky(h_p) == leaky(max_p h_p)
    // -> only 3 FMA + 1 max per point; one leaky per group.
    const float w0 = W_lift[lane], w1 = W_lift[32 + lane], w2 = W_lift[64 + lane];
    const float bl = b_lift[lane];
    // 8x8 block points reordered by ring group: inner(4) | inter(12) | outer(48)
    constexpr int ORD[64] = {
        27, 28, 35, 36,
        18, 19, 20, 21, 26, 29, 34, 37, 42, 43, 44, 45,
        0, 1, 2, 3, 4, 5, 6, 7, 8, 9, 10, 11, 12, 13, 14, 15,
        16, 17, 22, 23, 24, 25, 30, 31, 32, 33, 38, 39, 40, 41, 46, 47,
        48, 49, 50, 51, 52, 53, 54, 55, 56, 57, 58, 59, 60, 61, 62, 63};
    constexpr int GB[4] = {0, 4, 16, 64};

    for (int blk = 0; blk < 16; ++blk) {
        int b = blockBase + blk;
        int batch = b >> 10, g = b & 1023;
        int gi = g >> 5, gj = g & 31;
        int base = __builtin_amdgcn_readfirstlane(
            batch * 196608 + (gi * 8 * 256 + gj * 8) * 3);
        const float* p = pgi + base;
#pragma unroll
        for (int grp = 0; grp < 3; ++grp) {
            float acc = -3.0e38f;
#pragma unroll
            for (int q = GB[grp]; q < GB[grp + 1]; ++q) {
                int idx = ORD[q];
                int off = ((idx >> 3) * 256 + (idx & 7)) * 3;
                float x = p[off], y = p[off + 1], z = p[off + 2];
                float h = fmaf(x, w0, fmaf(y, w1, fmaf(z, w2, bl)));
                acc = fmaxf(acc, h);
            }
            acc = fmaxf(acc, 0.2f * acc);                      // leaky once per group
            bufA[blk * 96 + grp * 32 + lane] = (_Float16)acc;  // v-tile 16x96
        }
    }

    // ---- GEMM chain (WMMA f16 -> f32 acc), LDS ping-pong, no barriers ----
    mlp_layer<96, 128, 128, 1>(bufA, bufB, frag + FRAG_CODE, b_code, lane); // leaky
    mlp_layer<128, 64, 64, 0>(bufB, bufA, frag + FRAG_H1, b_h1, lane);      // relu
    mlp_layer<64, 32, 32, 0>(bufA, bufB, frag + FRAG_H2, b_h2, lane);       // relu
    mlp_layer<32, 16, 32, 0>(bufB, bufA, frag + FRAG_H3, b_h3, lane);       // relu, 16x32 tile

    // ---- final 16->3 layer: zero-pad K to 32, one WMMA, store cols 0..2 ----
    {
        const int hi = lane >> 4, ln = lane & 15;
#pragma unroll
        for (int g = 0; g < 8; ++g)
            bufA[(g + hi * 8) * 32 + 16 + ln] = (_Float16)0.0f;  // K-pad cols 16..31
        float bn = (ln < 3) ? b_h4[ln] : 0.0f;
        v8f c = {bn, bn, bn, bn, bn, bn, bn, bn};
        const _Float16* ap = bufA + ln * 32 + hi * 16;
        v8h a0 = *(const v8h*)ap;
        v8h a1 = *(const v8h*)(ap + 8);
        v16h a = __builtin_shufflevector(a0, a1, 0, 1, 2, 3, 4, 5, 6, 7,
                                         8, 9, 10, 11, 12, 13, 14, 15);
        v16h bb = *(const v16h*)(frag + FRAG_H4 + lane * 16);
        c = __builtin_amdgcn_wmma_f32_16x16x32_f16(false, a, false, bb,
                                                   (short)0, c, false, false);
        if (ln < 3) {
#pragma unroll
            for (int g = 0; g < 8; ++g)
                out[(blockBase + g + hi * 8) * 3 + ln] = c[g];
        }
    }
}

extern "C" void kernel_launch(void* const* d_in, const int* in_sizes, int n_in,
                              void* d_out, int out_size, void* d_ws, size_t ws_size,
                              hipStream_t stream) {
    const float* pgi = (const float*)d_in[0];
    const float* Wl  = (const float*)d_in[1];
    const float* bl  = (const float*)d_in[2];
    const float* Wc  = (const float*)d_in[3];
    const float* bc  = (const float*)d_in[4];
    const float* W1  = (const float*)d_in[5];
    const float* b1  = (const float*)d_in[6];
    const float* W2  = (const float*)d_in[7];
    const float* b2  = (const float*)d_in[8];
    const float* W3  = (const float*)d_in[9];
    const float* b3  = (const float*)d_in[10];
    const float* W4  = (const float*)d_in[11];
    const float* b4  = (const float*)d_in[12];
    _Float16* frag = (_Float16*)d_ws;  // 23552 halves = 46 KB

    int B = in_sizes[0] / (65536 * 3);           // batch from pgi size
    int tiles = (B * 1024) / 16;                 // 16 blocks per wave
    int wgs = tiles / WAVES_PER_WG;              // 4 waves per workgroup

    prep_weights<<<46, 512, 0, stream>>>(Wc, W1, W2, W3, W4, frag);
    flatnet_kernel<<<wgs, 128, 0, stream>>>(pgi, Wl, bl, bc, b1, b2, b3, b4,
                                            frag, (float*)d_out);
}
